// HtapFormer_26826365731318
// MI455X (gfx1250) — matrix operand order repaired and checked by reading, not compiled, not run on hardware
//
#include <hip/hip_runtime.h>

// ---------------- model dims ----------------
#define BB    256
#define NN    127
#define SS    128
#define HH    8
#define LL    8
#define EMB   64
#define HID   329        // EMB*5 + EMB/8 + 1
#define DH    41
#define HDH   328        // H*DH
#define FFN   512
#define PH    256
#define BINS  50
#define FEAT  1165
#define FD    73         // EMB + EMB/8 + 1
#define BN    (BB*NN)    // 32512
#define MS    (BB*SS)    // 32768
#define LAMF  0.1f

typedef __bf16  v16bf __attribute__((ext_vector_type(16)));
typedef __bf16  v8bf  __attribute__((ext_vector_type(8)));
typedef __bf16  v4bf  __attribute__((ext_vector_type(4)));
typedef float   v8f   __attribute__((ext_vector_type(8)));

__device__ __forceinline__ v16bf frag_cat(v8bf lo, v8bf hi) {
  return __builtin_shufflevector(lo, hi, 0,1,2,3,4,5,6,7,8,9,10,11,12,13,14,15);
}

__device__ __forceinline__ float apply_act(int act, float v) {
  switch (act) {
    case 1: return v > 0.f ? v : 0.01f * v;                           // leaky_relu
    case 2: return fmaxf(v, 0.f);                                     // relu
    case 3: return 0.5f * v * (1.f + erff(v * 0.70710678118654752f)); // exact gelu
    case 4: return 1.f / (1.f + __expf(-v));                          // sigmoid
  }
  return v;
}

// ---------------------------------------------------------------------------
// Generic tiled GEMM: C[m,n] = res + alpha*act(sum_k A[m,k]*B[k,n] + bias[n])
// bf16 WMMA, f32 accumulate. Tile 128x64, 8 waves. LDS layouts are
// fragment-native: A row-major (K contiguous), B n-major (transposed) so every
// WMMA fragment is two ds_load_b128's per lane.
// ---------------------------------------------------------------------------
__global__ __launch_bounds__(256) void gemm_wmma_kernel(
    float* __restrict__ C, int ldc,
    const float* __restrict__ A, int lda,
    const float* __restrict__ Bm, int ldb,
    const float* __restrict__ bias,
    const float* __restrict__ res, int ldres,
    int M, int N, int K, int act, float alpha)
{
  __shared__ __align__(16) __bf16 As[128][32];    // [m][k]
  __shared__ __align__(16) __bf16 Bst[64][32];    // [n][k]  (transposed)
  const int t = threadIdx.x;
  const int w = t >> 5, lane = t & 31, hs = lane >> 4, ml = lane & 15;
  const int m0 = blockIdx.y * 128, n0 = blockIdx.x * 64;

  v8f acc[4] = {};
  for (int k0 = 0; k0 < K; k0 += 32) {
    // prefetch next K-tile (global_prefetch_b8)
    if (k0 + 32 < K) {
      int pm = m0 + (t >> 1);
      if (pm < M) __builtin_prefetch(&A[(size_t)pm * lda + k0 + 32 + (t & 1) * 16], 0, 1);
      int pk = k0 + 32 + (t >> 3);
      if (pk < K && (t & 7) == 0) __builtin_prefetch(&Bm[(size_t)pk * ldb + n0], 0, 1);
    }
    // fill A: each thread 4 runs of 4 contiguous k -> ds_store_b64
    for (int i = t * 4; i < 128 * 32; i += 256 * 4) {
      int r = i >> 5, c = i & 31;
      int gm = m0 + r;
      v4bf pk;
#pragma unroll
      for (int j = 0; j < 4; ++j) {
        int gk = k0 + c + j;
        float v = (gm < M && gk < K) ? A[(size_t)gm * lda + gk] : 0.f;
        pk[j] = (__bf16)v;
      }
      *(v4bf*)&As[r][c] = pk;
    }
    // fill B transposed: thread owns (n, 4 consecutive k); global reads stay
    // coalesced over n within each j
    for (int i = t; i < 64 * 8; i += 256) {
      int n = i & 63, kc = (i >> 6) * 4;
      int gn = n0 + n;
      v4bf pk;
#pragma unroll
      for (int j = 0; j < 4; ++j) {
        int gk = k0 + kc + j;
        float v = (gk < K && gn < N) ? Bm[(size_t)gk * ldb + gn] : 0.f;
        pk[j] = (__bf16)v;
      }
      *(v4bf*)&Bst[n][kc] = pk;
    }
    __syncthreads();

    // A fragment: two contiguous 8-half runs (ISA 16-bit A layout)
    const v8bf* ar = (const v8bf*)&As[w * 16 + ml][0];
    v16bf af = frag_cat(ar[hs], ar[2 + hs]);
#pragma unroll
    for (int nt = 0; nt < 4; ++nt) {
      const v8bf* br = (const v8bf*)&Bst[nt * 16 + ml][0];
      v16bf bf = frag_cat(br[hs], br[2 + hs]);
      acc[nt] = __builtin_amdgcn_wmma_f32_16x16x32_bf16(
          false, af, false, bf, (short)0, acc[nt], false, false);
    }
    __syncthreads();
  }
  // epilogue
#pragma unroll
  for (int nt = 0; nt < 4; ++nt) {
#pragma unroll
    for (int r = 0; r < 8; ++r) {
      int m = m0 + w * 16 + hs * 8 + r;
      int n = n0 + nt * 16 + ml;
      if (m < M && n < N) {
        float v = acc[nt][r];
        if (bias) v += bias[n];
        v = apply_act(act, v);
        v *= alpha;
        if (res) v += res[(size_t)m * ldres + n];
        C[(size_t)m * ldc + n] = v;
      }
    }
  }
}

// ---------------------------------------------------------------------------
// Fused attention for one (b,h): scores = QK^T + bias, softmax, O = P @ V.
// Q/K row-major [row][d] (fragment-native for both A and B roles of QK^T),
// V transposed [d][j] (fragment-native B for P@V), P row-major [i][j].
// Q/K union-aliased with P.
// ---------------------------------------------------------------------------
__global__ __launch_bounds__(256) void attn_kernel(
    float* __restrict__ O, const float* __restrict__ Q,
    const float* __restrict__ Kp, const float* __restrict__ V,
    const float* __restrict__ biasTab)
{
  const int bh = blockIdx.x;
  const int b = bh >> 3, h = bh & 7;
  __shared__ __align__(16) union UQP {
    struct { __bf16 Q[128][64]; __bf16 K[128][64]; } qk;
    __bf16 P[128][128];
  } sm;
  __shared__ __align__(16) __bf16 Vst[64][128];   // [d][j]

  const int t = threadIdx.x;
  const int w = t >> 5, lane = t & 31, hs = lane >> 4, ml = lane & 15;

  // load Q,K (row-major, 4-half vector stores)
  for (int i = t; i < 128 * 16; i += 256) {
    int r = i >> 4, c4 = (i & 15) * 4;
    v4bf q4, k4;
#pragma unroll
    for (int j = 0; j < 4; ++j) {
      int d = c4 + j;
      float qv = 0.f, kv = 0.f;
      if (d < DH) {
        size_t base = ((size_t)(b * SS + r)) * HDH + h * DH + d;
        qv = Q[base]; kv = Kp[base];
      }
      q4[j] = (__bf16)qv; k4[j] = (__bf16)kv;
    }
    *(v4bf*)&sm.qk.Q[r][c4] = q4;
    *(v4bf*)&sm.qk.K[r][c4] = k4;
  }
  // load V transposed: thread owns (d, 4 consecutive rows)
  for (int i = t; i < 64 * 32; i += 256) {
    int d = i >> 5, r0 = (i & 31) * 4;
    v4bf v4;
#pragma unroll
    for (int j = 0; j < 4; ++j) {
      int r = r0 + j;
      float vv = (d < DH) ? V[((size_t)(b * SS + r)) * HDH + h * DH + d] : 0.f;
      v4[j] = (__bf16)vv;
    }
    *(v4bf*)&Vst[d][r0] = v4;
  }
  __syncthreads();

  const float* bp = biasTab + ((size_t)bh) * SS * SS;
  const int i0 = w * 16;

  // A fragments for this wave's rows (2 K-steps over padded d=64)
  const v8bf* qr = (const v8bf*)&sm.qk.Q[i0 + ml][0];   // 8 chunks per row
  v16bf aq[2];
#pragma unroll
  for (int ks = 0; ks < 2; ++ks)
    aq[ks] = frag_cat(qr[ks * 4 + hs], qr[ks * 4 + 2 + hs]);

  v8f sc[8];
#pragma unroll
  for (int tj = 0; tj < 8; ++tj) {
    v8f a = {};
    const v8bf* kr = (const v8bf*)&sm.qk.K[tj * 16 + ml][0];
#pragma unroll
    for (int ks = 0; ks < 2; ++ks) {
      v16bf bf = frag_cat(kr[ks * 4 + hs], kr[ks * 4 + 2 + hs]);
      a = __builtin_amdgcn_wmma_f32_16x16x32_bf16(
          false, aq[ks], false, bf, (short)0, a, false, false);
    }
#pragma unroll
    for (int r = 0; r < 8; ++r) {
      int i = i0 + hs * 8 + r, j = tj * 16 + ml;
      a[r] += bp[i * SS + j];
    }
    sc[tj] = a;
  }
  __syncthreads();   // all waves done reading Q/K before P overwrites them

  // row softmax: each row spread over 8 tiles (regs) x 16 lanes
#pragma unroll
  for (int r = 0; r < 8; ++r) {
    float mx = -3.0e38f;
#pragma unroll
    for (int tj = 0; tj < 8; ++tj) mx = fmaxf(mx, sc[tj][r]);
    for (int off = 8; off >= 1; off >>= 1) mx = fmaxf(mx, __shfl_xor(mx, off, 16));
    float ssum = 0.f;
#pragma unroll
    for (int tj = 0; tj < 8; ++tj) { float e = __expf(sc[tj][r] - mx); sc[tj][r] = e; ssum += e; }
    for (int off = 8; off >= 1; off >>= 1) ssum += __shfl_xor(ssum, off, 16);
    float inv = 1.f / ssum;
    int i = i0 + hs * 8 + r;
#pragma unroll
    for (int tj = 0; tj < 8; ++tj)
      sm.P[i][tj * 16 + ml] = (__bf16)(sc[tj][r] * inv);
  }
  __syncthreads();

  // O = P(128x128) @ V(128x64pad)
  v8f oc[4] = {};
  const v8bf* pr = (const v8bf*)&sm.P[i0 + ml][0];      // 16 chunks per row
#pragma unroll
  for (int ks = 0; ks < 4; ++ks) {
    v16bf ap = frag_cat(pr[ks * 4 + hs], pr[ks * 4 + 2 + hs]);
#pragma unroll
    for (int nt = 0; nt < 4; ++nt) {
      const v8bf* vr = (const v8bf*)&Vst[nt * 16 + ml][0];
      v16bf bv = frag_cat(vr[ks * 4 + hs], vr[ks * 4 + 2 + hs]);
      oc[nt] = __builtin_amdgcn_wmma_f32_16x16x32_bf16(
          false, ap, false, bv, (short)0, oc[nt], false, false);
    }
  }
#pragma unroll
  for (int nt = 0; nt < 4; ++nt)
#pragma unroll
    for (int r = 0; r < 8; ++r) {
      int i = i0 + hs * 8 + r, d = nt * 16 + ml;
      if (d < DH)
        O[((size_t)(b * SS + i)) * HDH + h * DH + d] = oc[nt][r];
    }
}

// ---------------------------------------------------------------------------
// LayerNorm: out[row, d] = (x-mu)*rsqrt(var+1e-5)*g[d] + b[d]; ld-strided rows
// ---------------------------------------------------------------------------
__global__ __launch_bounds__(128) void ln_kernel(
    float* __restrict__ out, long ldo, const float* __restrict__ in, long ldi,
    const float* __restrict__ g, const float* __restrict__ bta, int D)
{
  long row = blockIdx.x;
  const float* xr = in + row * ldi;
  __shared__ float s1[128], s2[128];
  float a = 0.f, b = 0.f;
  for (int d = threadIdx.x; d < D; d += 128) { float v = xr[d]; a += v; b += v * v; }
  s1[threadIdx.x] = a; s2[threadIdx.x] = b;
  __syncthreads();
  for (int off = 64; off > 0; off >>= 1) {
    if (threadIdx.x < off) { s1[threadIdx.x] += s1[threadIdx.x + off]; s2[threadIdx.x] += s2[threadIdx.x + off]; }
    __syncthreads();
  }
  float mu = s1[0] / D;
  float var = s2[0] / D - mu * mu;
  float inv = rsqrtf(var + 1e-5f);
  float* orow = out + row * ldo;
  for (int d = threadIdx.x; d < D; d += 128)
    orow[d] = (xr[d] - mu) * inv * g[d] + bta[d];
}

// ---------------------------------------------------------------------------
// Feature embedding: per-node filter MLP, hist proj, gathers -> final (BN,HID)
// ---------------------------------------------------------------------------
__global__ __launch_bounds__(256) void node_embed_kernel(
    float* __restrict__ final_, const float* __restrict__ x,
    const float* __restrict__ sampOut,
    const float* __restrict__ type_emb, const float* __restrict__ join_emb,
    const float* __restrict__ col_emb, const float* __restrict__ op_emb,
    const float* __restrict__ table_emb,
    const float* __restrict__ fW1, const float* __restrict__ fb1,
    const float* __restrict__ fW2, const float* __restrict__ fb2,
    const float* __restrict__ samp_b,
    const float* __restrict__ hist_W, const float* __restrict__ hist_b)
{
  const int n = blockIdx.x;
  const float* xr = x + (size_t)n * FEAT;
  const int t = threadIdx.x;
  __shared__ float ccA[3][FD], ccB[3][FD];
  __shared__ float msk[3];
  __shared__ float nf;
  if (t < 3) msk[t] = xr[11 + t];
  if (t == 0) nf = xr[11] + xr[12] + xr[13];

  if (t < 3 * FD) {
    int f = t / FD, d = t % FD;
    float v;
    if (d < EMB)      { int c = (int)xr[2 + f]; v = col_emb[c * EMB + d]; }
    else if (d < 72)  { int o = (int)xr[5 + f]; v = op_emb[o * 8 + (d - EMB)]; }
    else              v = xr[8 + f];
    ccA[f][d] = v;
  }
  __syncthreads();
  if (t < 3 * FD) {
    int f = t / FD, d = t % FD;
    float a = fb1[d];
    for (int k = 0; k < FD; ++k) a += ccA[f][k] * fW1[k * FD + d];
    ccB[f][d] = a > 0.f ? a : 0.01f * a;
  }
  __syncthreads();
  if (t < 3 * FD) {
    int f = t / FD, d = t % FD;
    float a = fb2[d];
    for (int k = 0; k < FD; ++k) a += ccB[f][k] * fW2[k * FD + d];
    ccA[f][d] = a > 0.f ? a : 0.01f * a;
  }
  __syncthreads();

  const int tid = (int)xr[0];
  const int jid = (int)xr[1];
  const int tb  = (int)xr[14 + BINS * 3];
  float* fr = final_ + (size_t)n * HID;
  for (int d = t; d < HID; d += 256) {
    float v;
    if (d < 64) {
      v = type_emb[tid * EMB + d];
    } else if (d < 137) {                       // filtEmb (FD=73)
      int dd = d - 64;
      v = (ccA[0][dd] * msk[0] + ccA[1][dd] * msk[1] + ccA[2][dd] * msk[2]) / nf;
    } else if (d < 201) {                       // joinEmb
      v = join_emb[jid * EMB + (d - 137)];
    } else if (d < 265) {                       // tabEmb
      int dd = d - 201;
      v = table_emb[tb * EMB + dd] + sampOut[(size_t)n * EMB + dd] + samp_b[dd];
    } else {                                    // histEmb
      int dd = d - 265;
      float a = 0.f;
      for (int f = 0; f < 3; ++f) {
        float s = hist_b[dd];
        for (int k = 0; k < BINS; ++k) s += xr[14 + k * 3 + f] * hist_W[k * EMB + dd];
        a += msk[f] * s;
      }
      v = a / nf;
    }
    fr[d] = v;
  }
}

// ---------------------------------------------------------------------------
// vs/vo head projections for the hb bias term
// ---------------------------------------------------------------------------
__global__ __launch_bounds__(256) void sproj_kernel(
    float* __restrict__ vsq, float* __restrict__ vsk,
    float* __restrict__ voq, float* __restrict__ vok,
    const int* __restrict__ storage_type, const int* __restrict__ node_operator,
    const float* __restrict__ counts,
    const float* __restrict__ st_emb, const float* __restrict__ cnt_W,
    const float* __restrict__ cnt_b, const float* __restrict__ nop_emb,
    const float* __restrict__ Wsq, const float* __restrict__ Wsk,
    const float* __restrict__ Woq, const float* __restrict__ Wok)
{
  int idx = blockIdx.x * 256 + threadIdx.x;
  if (idx >= BN) return;
  int st = storage_type[idx], no = node_operator[idx];
  float vs[8], vo[8];
  float c0 = counts[idx * 3 + 0], c1 = counts[idx * 3 + 1], c2 = counts[idx * 3 + 2];
#pragma unroll
  for (int d = 0; d < 8; ++d) {
    vs[d] = st_emb[st * 8 + d] + c0 * cnt_W[0 * 8 + d] + c1 * cnt_W[1 * 8 + d]
          + c2 * cnt_W[2 * 8 + d] + cnt_b[d];
    vo[d] = nop_emb[no * 8 + d];
  }
#pragma unroll
  for (int h = 0; h < 8; ++h) {
    float a = 0.f, b = 0.f, c = 0.f, e = 0.f;
#pragma unroll
    for (int d = 0; d < 8; ++d) {
      a += vs[d] * Wsq[d * 8 + h];
      b += vs[d] * Wsk[d * 8 + h];
      c += vo[d] * Woq[d * 8 + h];
      e += vo[d] * Wok[d * 8 + h];
    }
    vsq[idx * 8 + h] = a; vsk[idx * 8 + h] = b;
    voq[idx * 8 + h] = c; vok[idx * 8 + h] = e;
  }
}

// ---------------------------------------------------------------------------
// Precompute layer-invariant attention bias (B,H,S,S) — written once, read 8x
// ---------------------------------------------------------------------------
__global__ __launch_bounds__(256) void bias_kernel(
    float* __restrict__ biasTab, const float* __restrict__ attn_bias,
    const int* __restrict__ rel_pos, const float* __restrict__ rel_pos_emb,
    const float* __restrict__ super_dist,
    const float* __restrict__ vsq, const float* __restrict__ vsk,
    const float* __restrict__ voq, const float* __restrict__ vok)
{
  size_t idx = (size_t)blockIdx.x * 256 + threadIdx.x;
  const size_t total = (size_t)BB * HH * SS * SS;
  if (idx >= total) return;
  int j = idx % SS;
  int i = (idx / SS) % SS;
  int h = (idx / ((size_t)SS * SS)) % HH;
  int b = idx / ((size_t)SS * SS * HH);
  float v = attn_bias[((size_t)b * SS + i) * SS + j];
  if (i > 0 && j > 0) {
    int rp = rel_pos[((size_t)b * NN + (i - 1)) * NN + (j - 1)];
    v += rel_pos_emb[rp * HH + h];
    size_t qi = ((size_t)b * NN + (i - 1)) * 8 + h;
    size_t kj = ((size_t)b * NN + (j - 1)) * 8 + h;
    v += LAMF * (vsq[qi] * vsk[kj] + voq[qi] * vok[kj]);
  }
  float m = (i == 0) ? 1.f : ((j == 0) ? 1.f : 0.f);
  v += super_dist[h] * m;
  biasTab[idx] = v;
}

// ---------------------------------------------------------------------------
// h[b,0,:] = super_tok ; h[b,s,:] = nodeProj + height_emb[heights]
// ---------------------------------------------------------------------------
__global__ __launch_bounds__(256) void assemble_h_kernel(
    float* __restrict__ h, const float* __restrict__ nodeProj,
    const float* __restrict__ super_tok, const float* __restrict__ height_emb,
    const int* __restrict__ heights)
{
  size_t idx = (size_t)blockIdx.x * 256 + threadIdx.x;
  const size_t total = (size_t)MS * HID;
  if (idx >= total) return;
  int d = idx % HID;
  int s = (idx / HID) % SS;
  int b = idx / ((size_t)HID * SS);
  float v;
  if (s == 0) v = super_tok[d];
  else {
    size_t n = (size_t)b * NN + (s - 1);
    v = nodeProj[n * HID + d] + height_emb[(size_t)heights[n] * HID + d];
  }
  h[idx] = v;
}

// ---------------------------------------------------------------------------
static inline void gemm(hipStream_t stream, float* C, int ldc,
                        const float* A, int lda, const float* Bm, int ldb,
                        const float* bias, const float* res, int ldres,
                        int M, int N, int K, int act, float alpha)
{
  dim3 grid((N + 63) / 64, (M + 127) / 128);
  gemm_wmma_kernel<<<grid, 256, 0, stream>>>(C, ldc, A, lda, Bm, ldb, bias,
                                             res, ldres, M, N, K, act, alpha);
}

extern "C" void kernel_launch(void* const* d_in, const int* in_sizes, int n_in,
                              void* d_out, int out_size, void* d_ws, size_t ws_size,
                              hipStream_t stream)
{
  (void)in_sizes; (void)n_in; (void)out_size; (void)ws_size;
  int pi = 0;
  const float* x             = (const float*)d_in[pi++];
  const float* attn_bias_in  = (const float*)d_in[pi++];
  const int*   rel_pos       = (const int*)d_in[pi++];
  const int*   heights       = (const int*)d_in[pi++];
  const int*   storage_type  = (const int*)d_in[pi++];
  const int*   node_operator = (const int*)d_in[pi++];
  const float* counts        = (const float*)d_in[pi++];
  // params (insertion order)
  const float* type_emb   = (const float*)d_in[pi++];
  const float* join_emb   = (const float*)d_in[pi++];
  const float* col_emb    = (const float*)d_in[pi++];
  const float* op_emb_f   = (const float*)d_in[pi++];
  const float* table_emb  = (const float*)d_in[pi++];
  const float* filt_W1    = (const float*)d_in[pi++];
  const float* filt_b1    = (const float*)d_in[pi++];
  const float* filt_W2    = (const float*)d_in[pi++];
  const float* filt_b2    = (const float*)d_in[pi++];
  const float* samp_W     = (const float*)d_in[pi++];
  const float* samp_b     = (const float*)d_in[pi++];
  const float* hist_W     = (const float*)d_in[pi++];
  const float* hist_b     = (const float*)d_in[pi++];
  const float* proj_W     = (const float*)d_in[pi++];
  const float* proj_b     = (const float*)d_in[pi++];
  const float* rel_pos_emb= (const float*)d_in[pi++];
  const float* height_emb = (const float*)d_in[pi++];
  const float* super_tok  = (const float*)d_in[pi++];
  const float* super_dist = (const float*)d_in[pi++];
  const float* st_emb     = (const float*)d_in[pi++];
  const float* cnt_W      = (const float*)d_in[pi++];
  const float* cnt_b      = (const float*)d_in[pi++];
  const float* nop_emb    = (const float*)d_in[pi++];
  const float* Wsq        = (const float*)d_in[pi++];
  const float* Wsk        = (const float*)d_in[pi++];
  const float* Woq        = (const float*)d_in[pi++];
  const float* Wok        = (const float*)d_in[pi++];
  // layers (each leading dim L)
  const float* Lg1  = (const float*)d_in[pi++];
  const float* Lb1  = (const float*)d_in[pi++];
  const float* LWq  = (const float*)d_in[pi++];
  const float* Lbq  = (const float*)d_in[pi++];
  const float* LWk  = (const float*)d_in[pi++];
  const float* Lbk  = (const float*)d_in[pi++];
  const float* LWv  = (const float*)d_in[pi++];
  const float* Lbv  = (const float*)d_in[pi++];
  const float* LWo  = (const float*)d_in[pi++];
  const float* Lbo  = (const float*)d_in[pi++];
  const float* Lg2  = (const float*)d_in[pi++];
  const float* Lb2f = (const float*)d_in[pi++];
  const float* LF1  = (const float*)d_in[pi++];
  const float* Lc1  = (const float*)d_in[pi++];
  const float* LF2  = (const float*)d_in[pi++];
  const float* Lc2  = (const float*)d_in[pi++];
  const float* fin_g = (const float*)d_in[pi++];
  const float* fin_b = (const float*)d_in[pi++];
  const float* P1[8]; for (int i = 0; i < 8; ++i) P1[i] = (const float*)d_in[pi++];
  const float* P2[8]; for (int i = 0; i < 8; ++i) P2[i] = (const float*)d_in[pi++];

  // ---- workspace carve-out ----
  char* ws = (char*)d_ws;
  size_t off = 0;
  auto carve = [&](size_t bytes) -> float* {
    off = (off + 255) & ~(size_t)255;
    float* p = (float*)(ws + off);
    off += bytes;
    return p;
  };
  float* sampOut  = carve((size_t)BN * EMB * 4);
  float* finalBuf = carve((size_t)BN * HID * 4);
  float* nodeProj = carve((size_t)BN * HID * 4);
  float* hBuf     = carve((size_t)MS * HID * 4);
  float* biasTab  = carve((size_t)BB * HH * SS * SS * 4);
  float* vsq      = carve((size_t)BN * 8 * 4);
  float* vsk      = carve((size_t)BN * 8 * 4);
  float* voq      = carve((size_t)BN * 8 * 4);
  float* vok      = carve((size_t)BN * 8 * 4);
  float* yBuf     = carve((size_t)MS * HID * 4);
  float* qBuf     = carve((size_t)MS * HDH * 4);
  float* kBuf     = carve((size_t)MS * HDH * 4);
  float* vBuf     = carve((size_t)MS * HDH * 4);
  float* oBuf     = carve((size_t)MS * HDH * 4);
  float* midBuf   = carve((size_t)MS * FFN * 4);
  float* clsBuf   = carve((size_t)BB * HID * 4);
  float* pHid     = carve((size_t)BB * PH * 4);
  float* pMid     = carve((size_t)BB * PH * 4);
  float* pSum     = carve((size_t)BB * PH * 4);

  const float scaleQ = 0.15617376188860607f;   // 41^-0.5

  // ---- front-end ----
  // sampOut = x[:, 165:1165] @ samp_W   (samp_b added in node_embed)
  gemm(stream, sampOut, EMB, x + (15 + BINS * 3), FEAT, samp_W, EMB,
       nullptr, nullptr, 0, BN, EMB, 1000, 0, 1.f);
  node_embed_kernel<<<BN, 256, 0, stream>>>(finalBuf, x, sampOut,
      type_emb, join_emb, col_emb, op_emb_f, table_emb,
      filt_W1, filt_b1, filt_W2, filt_b2, samp_b, hist_W, hist_b);
  // nodeProj = leaky_relu(final @ proj_W + proj_b)
  gemm(stream, nodeProj, HID, finalBuf, HID, proj_W, HID, proj_b,
       nullptr, 0, BN, HID, HID, 1, 1.f);
  sproj_kernel<<<(BN + 255) / 256, 256, 0, stream>>>(vsq, vsk, voq, vok,
      storage_type, node_operator, counts, st_emb, cnt_W, cnt_b, nop_emb,
      Wsq, Wsk, Woq, Wok);
  assemble_h_kernel<<<((size_t)MS * HID + 255) / 256, 256, 0, stream>>>(
      hBuf, nodeProj, super_tok, height_emb, heights);
  bias_kernel<<<((size_t)BB * HH * SS * SS + 255) / 256, 256, 0, stream>>>(
      biasTab, attn_bias_in, rel_pos, rel_pos_emb, super_dist, vsq, vsk, voq, vok);

  // ---- transformer layers ----
  for (int l = 0; l < LL; ++l) {
    const float* g1  = Lg1  + (size_t)l * HID;
    const float* b1  = Lb1  + (size_t)l * HID;
    const float* Wq  = LWq  + (size_t)l * HID * HDH;
    const float* bq  = Lbq  + (size_t)l * HDH;
    const float* Wk  = LWk  + (size_t)l * HID * HDH;
    const float* bk  = Lbk  + (size_t)l * HDH;
    const float* Wv  = LWv  + (size_t)l * HID * HDH;
    const float* bv  = Lbv  + (size_t)l * HDH;
    const float* Wo  = LWo  + (size_t)l * HDH * HID;
    const float* bo  = Lbo  + (size_t)l * HID;
    const float* g2  = Lg2  + (size_t)l * HID;
    const float* b2f = Lb2f + (size_t)l * HID;
    const float* F1  = LF1  + (size_t)l * HID * FFN;
    const float* c1  = Lc1  + (size_t)l * FFN;
    const float* F2  = LF2  + (size_t)l * FFN * HID;
    const float* c2  = Lc2  + (size_t)l * HID;

    ln_kernel<<<MS, 128, 0, stream>>>(yBuf, HID, hBuf, HID, g1, b1, HID);
    gemm(stream, qBuf, HDH, yBuf, HID, Wq, HDH, bq, nullptr, 0, MS, HDH, HID, 0, scaleQ);
    gemm(stream, kBuf, HDH, yBuf, HID, Wk, HDH, bk, nullptr, 0, MS, HDH, HID, 0, 1.f);
    gemm(stream, vBuf, HDH, yBuf, HID, Wv, HDH, bv, nullptr, 0, MS, HDH, HID, 0, 1.f);
    attn_kernel<<<BB * HH, 256, 0, stream>>>(oBuf, qBuf, kBuf, vBuf, biasTab);
    // h = h + o @ Wo + bo   (in-place residual)
    gemm(stream, hBuf, HID, oBuf, HDH, Wo, HID, bo, hBuf, HID, MS, HID, HDH, 0, 1.f);
    ln_kernel<<<MS, 128, 0, stream>>>(yBuf, HID, hBuf, HID, g2, b2f, HID);
    gemm(stream, midBuf, FFN, yBuf, HID, F1, FFN, c1, nullptr, 0, MS, FFN, HID, 3, 1.f);
    gemm(stream, hBuf, HID, midBuf, FFN, F2, HID, c2, hBuf, HID, MS, HID, FFN, 0, 1.f);
  }

  // ---- final LN on cls rows only (row stride S*HID) ----
  ln_kernel<<<BB, 128, 0, stream>>>(clsBuf, HID, hBuf, (long)SS * HID, fin_g, fin_b, HID);

  // ---- prediction heads (write straight into d_out with ldc=2) ----
  float* out = (float*)d_out;
  for (int ph = 0; ph < 2; ++ph) {
    const float** P = (ph == 0) ? P1 : P2;
    gemm(stream, pHid, PH, clsBuf, HID, P[0], PH, P[1], nullptr, 0, BB, PH, HID, 2, 1.f);
    gemm(stream, pMid, PH, pHid, PH, P[2], PH, P[3], nullptr, 0, BB, PH, PH, 2, 1.f);
    gemm(stream, pSum, PH, pMid, PH, P[4], PH, P[5], pHid, PH, BB, PH, PH, 2, 1.f);
    gemm(stream, out + ph, 2, pSum, PH, P[6], 1, P[7], nullptr, 0, BB, 1, PH, 4, 1.f);
  }
}
